// MultiHeadSelfAttention_89498528514289
// MI455X (gfx1250) — compile-verified, matrix operand order checked by reading, not run on hardware
//
#include <hip/hip_runtime.h>
#include <hip/hip_bf16.h>
#include <math.h>

// Problem constants (B,S,D,H from reference)
#define B_   2
#define S_   2048
#define D_   1024
#define H_   16
#define A_   64
#define BS_  (B_ * S_)      // 4096 tokens
#define D3_  (3 * D_)       // 3072
#define EPSV 1e-3f
#define SCALE 0.125f                 // 1/sqrt(64)
#define NEGS (-100000.0f * 0.125f)   // masked score after /sqrt(A)

typedef __bf16 bf16_t;
typedef bf16_t v16bf __attribute__((ext_vector_type(16)));
typedef bf16_t v8bf  __attribute__((ext_vector_type(8)));
typedef float  v8f   __attribute__((ext_vector_type(8)));

// ---------------------------------------------------------------------------
// WMMA helpers
// ---------------------------------------------------------------------------
__device__ __forceinline__ v8f wmma_bf16(v16bf a, v16bf b, v8f c) {
  // D(16x16 f32) = A(16x32 bf16) * B(32x16 bf16) + C
  return __builtin_amdgcn_wmma_f32_16x16x32_bf16(
      /*neg_a=*/false, a, /*neg_b=*/false, b,
      /*c_mod=*/(short)0, c, /*reuse_a=*/false, /*reuse_b=*/false);
}

__device__ __forceinline__ v8f zero8() {
  v8f z;
#pragma unroll
  for (int i = 0; i < 8; ++i) z[i] = 0.0f;
  return z;
}

// Load a 16x32 bf16 fragment (A layout; also used for B given B^T row-major).
// Per ISA 7.12.2: lane&15 selects row; lanes>=16 hold K+8/K+24 halves.
__device__ __forceinline__ v16bf frag_ld16(const bf16_t* p) {
  v8bf lo = *reinterpret_cast<const v8bf*>(p);
  v8bf hi = *reinterpret_cast<const v8bf*>(p + 16);
  v16bf f;
#pragma unroll
  for (int i = 0; i < 8; ++i) { f[i] = lo[i]; f[i + 8] = hi[i]; }
  return f;
}

__device__ __forceinline__ v16bf frag_g(const bf16_t* __restrict__ base, int ld,
                                        int row0, int k0, int lane) {
  int r = row0 + (lane & 15);
  int koff = (lane & 16) ? 8 : 0;
  return frag_ld16(base + (size_t)r * (size_t)ld + (size_t)(k0 + koff));
}

// LDS fragment load (row stride in elements; multiple of 8 keeps 16B align)
__device__ __forceinline__ v16bf frag_lds(const bf16_t* p, int ld, int lane) {
  int r = lane & 15;
  int koff = (lane & 16) ? 8 : 0;
  return frag_ld16(p + r * ld + koff);
}

// 16-lane (half-wave) reductions: xor masks 1,2,4,8 stay inside each 16-lane
// group, matching the C-matrix row split (rows 0-7 lanes 0-15, rows 8-15
// lanes 16-31).
__device__ __forceinline__ float hmax16(float v) {
#pragma unroll
  for (int m = 1; m <= 8; m <<= 1) v = fmaxf(v, __shfl_xor(v, m, 32));
  return v;
}
__device__ __forceinline__ float hsum16(float v) {
#pragma unroll
  for (int m = 1; m <= 8; m <<= 1) v += __shfl_xor(v, m, 32);
  return v;
}

// ---------------------------------------------------------------------------
// Kernel 1: fp32 -> bf16 elementwise convert
// ---------------------------------------------------------------------------
__global__ void k_cvt(const float* __restrict__ x, bf16_t* __restrict__ y, int n) {
  int i = blockIdx.x * blockDim.x + threadIdx.x;
  if (i < n) y[i] = (bf16_t)x[i];
}

// ---------------------------------------------------------------------------
// Kernel 2: transpose + convert: w[rows][cols] fp32 -> wt[cols][rows] bf16
// ---------------------------------------------------------------------------
__global__ void k_transpose_cvt(const float* __restrict__ w,
                                bf16_t* __restrict__ wt,
                                int rows, int cols) {
  __shared__ float tile[32][33];
  int c0 = blockIdx.x * 32, r0 = blockIdx.y * 32;
  int tx = threadIdx.x, ty = threadIdx.y;  // block (32,8)
  for (int i = ty; i < 32; i += 8)
    tile[i][tx] = w[(size_t)(r0 + i) * cols + (c0 + tx)];
  __syncthreads();
  for (int i = ty; i < 32; i += 8)
    wt[(size_t)(c0 + i) * rows + (r0 + tx)] = (bf16_t)tile[tx][i];
}

// ---------------------------------------------------------------------------
// Kernel 3: QKV GEMM  [4096 x 1024] * [1024 x 3072] + bias
// grid (128 m-tiles, 6 n-groups), 256 threads = 8 waves, wave tile 32x64:
// 2 A-frags x 4 B-frags -> 8 wmma per 6 fragment loads.
// Writes Q,K as bf16 [B,H,S,A], V transposed as bf16 [B,H,A,S].
// ---------------------------------------------------------------------------
__global__ void __launch_bounds__(256) k_qkv(
    const bf16_t* __restrict__ xb, const bf16_t* __restrict__ wqkvT,
    const float* __restrict__ bqkv, bf16_t* __restrict__ qb,
    bf16_t* __restrict__ kb, bf16_t* __restrict__ vtb) {
  int lane = threadIdx.x & 31;
  int w = threadIdx.x >> 5;                 // 0..7
  int m0 = blockIdx.x * 32;                 // token tile (32 rows)
  int n0 = blockIdx.y * 512 + w * 64;       // output column base

  v8f acc[2][4];
#pragma unroll
  for (int t = 0; t < 2; ++t)
#pragma unroll
    for (int j = 0; j < 4; ++j) acc[t][j] = zero8();

  for (int k0 = 0; k0 < D_; k0 += 32) {
    v16bf a0 = frag_g(xb, D_, m0, k0, lane);
    v16bf a1 = frag_g(xb, D_, m0 + 16, k0, lane);
#pragma unroll
    for (int j = 0; j < 4; ++j) {
      v16bf bfr = frag_g(wqkvT, D_, n0 + 16 * j, k0, lane);
      acc[0][j] = wmma_bf16(a0, bfr, acc[0][j]);
      acc[1][j] = wmma_bf16(a1, bfr, acc[1][j]);
    }
  }

  int hi = (lane & 16) ? 8 : 0;
  int cl = lane & 15;
#pragma unroll
  for (int t = 0; t < 2; ++t) {
#pragma unroll
    for (int j = 0; j < 4; ++j) {
      int col = n0 + 16 * j + cl;
      float bias = bqkv[col];
#pragma unroll
      for (int r = 0; r < 8; ++r) {
        int tok = m0 + 16 * t + r + hi;
        int b = tok >> 11;            // / S_
        int s = tok & (S_ - 1);
        float v = acc[t][j][r] + bias;
        if (col < D_) {               // Q
          int h = col >> 6, a = col & 63;
          qb[(((size_t)(b * H_ + h)) * S_ + s) * A_ + a] = (bf16_t)v;
        } else if (col < 2 * D_) {    // K
          int c = col - D_;
          int h = c >> 6, a = c & 63;
          kb[(((size_t)(b * H_ + h)) * S_ + s) * A_ + a] = (bf16_t)v;
        } else {                      // V (transposed: [B,H,A,S])
          int c = col - 2 * D_;
          int h = c >> 6, a = c & 63;
          vtb[(((size_t)(b * H_ + h)) * A_ + a) * S_ + s] = (bf16_t)v;
        }
      }
    }
  }
}

// ---------------------------------------------------------------------------
// Kernel 4: flash attention.  grid (S/128, H, B), 128 threads = 4 waves,
// each wave owns 32 query rows (two 16-row tiles); key blocks of 32.
// K/V fragments are loaded once per key block and reused by both query
// tiles: 16 wmma per 8 fragment loads.  Online softmax in fp32.
// ---------------------------------------------------------------------------
__global__ void __launch_bounds__(128) k_attn(
    const bf16_t* __restrict__ qb, const bf16_t* __restrict__ kb,
    const bf16_t* __restrict__ vtb, const int* __restrict__ mask,
    bf16_t* __restrict__ ctx) {
  // per-wave, per-tile P staging; stride 40 bf16 = 80B keeps 16B-aligned
  // b128 ds loads and spreads the 16 rows over distinct bank groups.
  __shared__ __align__(16) bf16_t ldsP[4][2][16][40];

  int lane = threadIdx.x & 31;
  int w = threadIdx.x >> 5;
  int q0 = blockIdx.x * 128 + w * 32;       // query row base (within S)
  int h = blockIdx.y, b = blockIdx.z;
  size_t bh = (size_t)(b * H_ + h);
  const bf16_t* qbase = qb + bh * (size_t)S_ * A_;
  const bf16_t* kbase = kb + bh * (size_t)S_ * A_;
  const bf16_t* vbase = vtb + bh * (size_t)A_ * S_;
  const int* mbase = mask + (size_t)b * S_ * S_;

  // Q fragments resident for the whole pass: [tile][k-half]
  v16bf qa[2][2];
#pragma unroll
  for (int t = 0; t < 2; ++t) {
    qa[t][0] = frag_g(qbase, A_, q0 + 16 * t, 0, lane);
    qa[t][1] = frag_g(qbase, A_, q0 + 16 * t, 32, lane);
  }

  v8f acc[2][4];
#pragma unroll
  for (int t = 0; t < 2; ++t)
#pragma unroll
    for (int j = 0; j < 4; ++j) acc[t][j] = zero8();
  float rmax[2][8], rsum[2][8];
#pragma unroll
  for (int t = 0; t < 2; ++t)
#pragma unroll
    for (int r = 0; r < 8; ++r) {
      rmax[t][r] = -__builtin_inff();
      rsum[t][r] = 0.f;
    }

  int hi = (lane & 16) ? 8 : 0;
  int cl = lane & 15;

  for (int n0 = 0; n0 < S_; n0 += 32) {
    // K fragments: [key tile][k-half], shared by both query tiles
    v16bf kf[2][2];
    kf[0][0] = frag_g(kbase, A_, n0, 0, lane);
    kf[0][1] = frag_g(kbase, A_, n0, 32, lane);
    kf[1][0] = frag_g(kbase, A_, n0 + 16, 0, lane);
    kf[1][1] = frag_g(kbase, A_, n0 + 16, 32, lane);

    int c0 = n0 + cl, c1 = c0 + 16;
#pragma unroll
    for (int t = 0; t < 2; ++t) {
      v8f sc0 = zero8(), sc1 = zero8();
      sc0 = wmma_bf16(qa[t][0], kf[0][0], sc0);
      sc0 = wmma_bf16(qa[t][1], kf[0][1], sc0);
      sc1 = wmma_bf16(qa[t][0], kf[1][0], sc1);
      sc1 = wmma_bf16(qa[t][1], kf[1][1], sc1);

#pragma unroll
      for (int r = 0; r < 8; ++r) {
        int qrow = q0 + 16 * t + r + hi;
        const int* mrow = mbase + (size_t)qrow * S_;
        float s0 = mrow[c0] ? sc0[r] * SCALE : NEGS;
        float s1v = mrow[c1] ? sc1[r] * SCALE : NEGS;
        float tmax = hmax16(fmaxf(s0, s1v));
        float mnew = fmaxf(rmax[t][r], tmax);
        float f = __expf(rmax[t][r] - mnew);
        float p0 = __expf(s0 - mnew);
        float p1 = __expf(s1v - mnew);
        float tsum = hsum16(p0 + p1);
        rsum[t][r] = rsum[t][r] * f + tsum;
        rmax[t][r] = mnew;
#pragma unroll
        for (int j = 0; j < 4; ++j) acc[t][j][r] *= f;
        // stage P (C-layout lane=col) into LDS for A-layout reload
        ldsP[w][t][r + hi][cl] = (bf16_t)p0;
        ldsP[w][t][r + hi][cl + 16] = (bf16_t)p1;
      }
    }
    __syncthreads();  // order LDS writes before cross-lane fragment reads

    v16bf pf0 = frag_lds(&ldsP[w][0][0][0], 40, lane);
    v16bf pf1 = frag_lds(&ldsP[w][1][0][0], 40, lane);
#pragma unroll
    for (int j = 0; j < 4; ++j) {
      v16bf vf = frag_g(vbase, S_, j * 16, n0, lane);  // V^T rows = head dim
      acc[0][j] = wmma_bf16(pf0, vf, acc[0][j]);
      acc[1][j] = wmma_bf16(pf1, vf, acc[1][j]);
    }
    __syncthreads();
  }

  // normalize and emit context as bf16 [token][D]
#pragma unroll
  for (int t = 0; t < 2; ++t) {
#pragma unroll
    for (int j = 0; j < 4; ++j) {
      int d = h * A_ + j * 16 + cl;
#pragma unroll
      for (int r = 0; r < 8; ++r) {
        int tok = b * S_ + q0 + 16 * t + r + hi;
        float v = acc[t][j][r] / rsum[t][r];
        ctx[(size_t)tok * D_ + d] = (bf16_t)v;
      }
    }
  }
}

// ---------------------------------------------------------------------------
// Kernel 5: output projection + bias + residual + LayerNorm.
// grid 128 (32-token tiles), 256 threads = 8 waves, wave tile 32x128:
// 2 A-frags x 8 B-frags -> 16 wmma per 10 fragment loads.
// ---------------------------------------------------------------------------
__global__ void __launch_bounds__(256) k_out(
    const bf16_t* __restrict__ ctx, const bf16_t* __restrict__ woutT,
    const float* __restrict__ bout, const float* __restrict__ emb,
    const float* __restrict__ gamma, const float* __restrict__ beta,
    float* __restrict__ out) {
  __shared__ float redS[8][32], redQ[8][32], muA[32], rstdA[32];

  int lane = threadIdx.x & 31;
  int w = threadIdx.x >> 5;                 // 0..7
  int m0 = blockIdx.x * 32;

  v8f acc[2][8];
#pragma unroll
  for (int t = 0; t < 2; ++t)
#pragma unroll
    for (int j = 0; j < 8; ++j) acc[t][j] = zero8();

  for (int k0 = 0; k0 < D_; k0 += 32) {
    v16bf a0 = frag_g(ctx, D_, m0, k0, lane);
    v16bf a1 = frag_g(ctx, D_, m0 + 16, k0, lane);
#pragma unroll
    for (int j = 0; j < 8; ++j) {
      v16bf bfr = frag_g(woutT, D_, w * 128 + j * 16, k0, lane);
      acc[0][j] = wmma_bf16(a0, bfr, acc[0][j]);
      acc[1][j] = wmma_bf16(a1, bfr, acc[1][j]);
    }
  }

  int hi = (lane & 16) ? 8 : 0;
  int cl = lane & 15;
  float s1[2][8], s2[2][8];
#pragma unroll
  for (int t = 0; t < 2; ++t)
#pragma unroll
    for (int r = 0; r < 8; ++r) { s1[t][r] = 0.f; s2[t][r] = 0.f; }

#pragma unroll
  for (int t = 0; t < 2; ++t) {
#pragma unroll
    for (int j = 0; j < 8; ++j) {
      int col = w * 128 + j * 16 + cl;
      float bias = bout[col];
#pragma unroll
      for (int r = 0; r < 8; ++r) {
        int tok = m0 + 16 * t + r + hi;
        float y = acc[t][j][r] + bias + emb[(size_t)tok * D_ + col];
        acc[t][j][r] = y;
        s1[t][r] += y;
        s2[t][r] += y * y;
      }
    }
  }
#pragma unroll
  for (int t = 0; t < 2; ++t)
#pragma unroll
    for (int r = 0; r < 8; ++r) {
      s1[t][r] = hsum16(s1[t][r]);
      s2[t][r] = hsum16(s2[t][r]);
    }
  if (cl == 0) {
#pragma unroll
    for (int t = 0; t < 2; ++t)
#pragma unroll
      for (int r = 0; r < 8; ++r) {
        redS[w][16 * t + r + hi] = s1[t][r];
        redQ[w][16 * t + r + hi] = s2[t][r];
      }
  }
  __syncthreads();
  if (threadIdx.x < 32) {
    int row = threadIdx.x;
    float s = 0.f, q = 0.f;
#pragma unroll
    for (int i = 0; i < 8; ++i) { s += redS[i][row]; q += redQ[i][row]; }
    float mu = s * (1.0f / D_);
    float var = q * (1.0f / D_) - mu * mu;
    muA[row] = mu;
    rstdA[row] = rsqrtf(var + EPSV);
  }
  __syncthreads();

#pragma unroll
  for (int t = 0; t < 2; ++t) {
#pragma unroll
    for (int j = 0; j < 8; ++j) {
      int col = w * 128 + j * 16 + cl;
      float g = gamma[col], be = beta[col];
#pragma unroll
      for (int r = 0; r < 8; ++r) {
        int row = 16 * t + r + hi;
        int tok = m0 + row;
        out[(size_t)tok * D_ + col] =
            (acc[t][j][r] - muA[row]) * rstdA[row] * g + be;
      }
    }
  }
}

// ---------------------------------------------------------------------------
// Launch
// ---------------------------------------------------------------------------
extern "C" void kernel_launch(void* const* d_in, const int* in_sizes, int n_in,
                              void* d_out, int out_size, void* d_ws, size_t ws_size,
                              hipStream_t stream) {
  (void)in_sizes; (void)n_in; (void)out_size; (void)ws_size;
  const float* emb   = (const float*)d_in[0];
  const int*   mask  = (const int*)d_in[1];
  const float* wqkv  = (const float*)d_in[2];
  const float* bqkv  = (const float*)d_in[3];
  const float* wout  = (const float*)d_in[4];
  const float* bout  = (const float*)d_in[5];
  const float* gamma = (const float*)d_in[6];
  const float* beta  = (const float*)d_in[7];
  float* out = (float*)d_out;

  // workspace carve-out (all 256B-aligned); total ~50.3 MB bf16 scratch
  char* ws = (char*)d_ws;
  size_t o = 0;
  auto carve = [&](size_t bytes) -> char* {
    o = (o + 255) & ~(size_t)255;
    char* p = ws + o;
    o += bytes;
    return p;
  };
  bf16_t* xb     = (bf16_t*)carve((size_t)BS_ * D_ * 2);
  bf16_t* wqkvT  = (bf16_t*)carve((size_t)D3_ * D_ * 2);
  bf16_t* woutT  = (bf16_t*)carve((size_t)D_ * D_ * 2);
  bf16_t* qb     = (bf16_t*)carve((size_t)BS_ * D_ * 2);
  bf16_t* kb     = (bf16_t*)carve((size_t)BS_ * D_ * 2);
  bf16_t* vtb    = (bf16_t*)carve((size_t)BS_ * D_ * 2);
  bf16_t* ctxb   = (bf16_t*)carve((size_t)BS_ * D_ * 2);

  // 1) precision conversion + weight transposes
  {
    int n = BS_ * D_;
    k_cvt<<<(n + 255) / 256, 256, 0, stream>>>(emb, xb, n);
  }
  k_transpose_cvt<<<dim3(D3_ / 32, D_ / 32), dim3(32, 8), 0, stream>>>(
      wqkv, wqkvT, D_, D3_);
  k_transpose_cvt<<<dim3(D_ / 32, D_ / 32), dim3(32, 8), 0, stream>>>(
      wout, woutT, D_, D_);

  // 2) QKV projection
  k_qkv<<<dim3(BS_ / 32, D3_ / 512), 256, 0, stream>>>(xb, wqkvT, bqkv, qb, kb, vtb);

  // 3) flash attention
  k_attn<<<dim3(S_ / 128, H_, B_), 128, 0, stream>>>(qb, kb, vtb, mask, ctxb);

  // 4) output projection + residual + layernorm
  k_out<<<dim3(BS_ / 32), 256, 0, stream>>>(ctxb, woutT, bout, emb, gamma, beta, out);
}
